// FeedForwardQuantum_65481071397135
// MI455X (gfx1250) — matrix operand order repaired and checked by reading, not compiled, not run on hardware
//
#include <hip/hip_runtime.h>

typedef __attribute__((ext_vector_type(8)))  __bf16   v8bf;
typedef __attribute__((ext_vector_type(16))) __bf16   v16bf;
typedef __attribute__((ext_vector_type(8)))  float    v8f;
typedef __attribute__((ext_vector_type(4)))  unsigned v4u;
typedef __attribute__((ext_vector_type(8)))  unsigned v8u;

#define E_DIM    512
#define FFN_DIM  2048
#define NQ_DIM   8
#define M_ROWS   128
#define THREADS  512
#define K_STEP   32
#define NSTEPS   (FFN_DIM / K_STEP)   // 64
#define HPAD     40   // LDS slab row stride (bf16): 64B data + 16B pad -> conflict-free b128

__device__ __forceinline__ unsigned short f2bf_rne(float f) {
  unsigned u = __builtin_bit_cast(unsigned, f);
  u += 0x7FFFu + ((u >> 16) & 1u);          // round-to-nearest-even
  return (unsigned short)(u >> 16);
}

// One-time conversion of W2 (f32 [512][2048]) -> bf16 row-major copy in workspace.
__global__ void cvt_w2_bf16(const float* __restrict__ w2,
                            unsigned short* __restrict__ w2b, int n4) {
  int i = blockIdx.x * blockDim.x + threadIdx.x;
  if (i < n4) {
    const float4 v = reinterpret_cast<const float4*>(w2)[i];
    ushort4 r;
    r.x = f2bf_rne(v.x); r.y = f2bf_rne(v.y);
    r.z = f2bf_rne(v.z); r.w = f2bf_rne(v.w);
    reinterpret_cast<ushort4*>(w2b)[i] = r;
  }
}

// Low 32 bits of a generic pointer to __shared__ == LDS offset (ISA 10.2 aperture map).
__device__ __forceinline__ unsigned lds_off(const void* p) {
  return (unsigned)(unsigned long long)p;
}

// Tensor Data Mover: DMA one 512x32 bf16 W2 slab (2D tile) into LDS with HW padding
// (pad 4 DWORDs after every 16 DWORDs -> 80B LDS row stride). Tracked by TENSORcnt.
__device__ __forceinline__ void tdm_stage_b(unsigned lds_base, unsigned long long gaddr) {
  v4u g0;
  g0.x = 1u;                                                      // count=1, user D#
  g0.y = lds_base;                                                // lds_addr
  g0.z = (unsigned)gaddr;                                         // global_addr[31:0]
  g0.w = (unsigned)((gaddr >> 32) & 0x01FFFFFFull) | (2u << 30);  // [56:32] | type=2
  v8u g1;
  g1.s0 = (1u << 16)        // data_size = 1 -> 2 bytes
        | (1u << 20)        // pad_enable
        | (3u << 22)        // pad_interval = 3 -> every 16 DWORDs (64B)
        | (3u << 25);       // pad_amount   = 3 -> 4 DWORDs (16B)
  g1.s1 = (unsigned)(FFN_DIM & 0xFFFF) << 16;   // tensor_dim0[15:0]   (2048)
  g1.s2 = (unsigned)(E_DIM & 0xFFFF) << 16;     // tensor_dim1[15:0]   (512)
  g1.s3 = (unsigned)K_STEP << 16;               // tile_dim0           (32)
  g1.s4 = (unsigned)E_DIM;                      // tile_dim1           (512)
  g1.s5 = (unsigned)FFN_DIM;                    // tensor_dim0_stride  (2048)
  g1.s6 = 0u;
  g1.s7 = 0u;
  asm volatile("tensor_load_to_lds %0, %1" :: "s"(g0), "s"(g1) : "memory");
}

// Fused: meas = cos(x[:, :8] + rxs); h-slab = relu(meas @ W1^T) -> LDS(bf16);
// out = h @ W2^T via v_wmma_f32_16x16x32_bf16; B slabs staged by the TDM,
// double-buffered K pipeline; B register fetch pipelined exactly one tile ahead,
// with sched_barriers capping in-flight LDS loads (no spills).
__launch_bounds__(THREADS, 1)
__global__ void ffq_fused(const float* __restrict__ x,
                          const float* __restrict__ rxs,
                          const float* __restrict__ w1,
                          const unsigned short* __restrict__ w2b,
                          float* __restrict__ out) {
  __shared__ float smeas[M_ROWS][NQ_DIM];                        //  4 KB
  __shared__ __align__(16) unsigned short sh_h[2][M_ROWS][HPAD]; // 20.5 KB
  __shared__ __align__(16) unsigned short sb[2][E_DIM][HPAD];    // 82 KB

  const int tid  = threadIdx.x;
  const int wv   = tid >> 5;
  const int row0 = blockIdx.x * M_ROWS;

  // ---- measurements: only the first 8 of 512 features per row are used
  for (int i = tid; i < M_ROWS * NQ_DIM; i += THREADS) {
    int m = i >> 3, q = i & 7;
    smeas[m][q] = __builtin_cosf(x[(size_t)(row0 + m) * E_DIM + q] + rxs[q]);
  }
  __syncthreads();

  // Producer: h[:, k1:k1+32] = relu(meas @ W1[k1:k1+32]^T) -> bf16 LDS slab.
  auto compute_h = [&](int k1, int buf) {
    const int fl = tid & 31;            // local FFN column
    const int mg = tid >> 5;            // 16 row-groups of 8
    const float4 wa = reinterpret_cast<const float4*>(w1 + (size_t)(k1 + fl) * NQ_DIM)[0];
    const float4 wb = reinterpret_cast<const float4*>(w1 + (size_t)(k1 + fl) * NQ_DIM)[1];
    #pragma unroll
    for (int j = 0; j < 8; ++j) {
      const int m = mg * 8 + j;
      float s = smeas[m][0]*wa.x + smeas[m][1]*wa.y + smeas[m][2]*wa.z + smeas[m][3]*wa.w
              + smeas[m][4]*wb.x + smeas[m][5]*wb.y + smeas[m][6]*wb.z + smeas[m][7]*wb.w;
      sh_h[buf][m][fl] = f2bf_rne(fmaxf(s, 0.0f));
    }
  };

  // Producer: one TDM descriptor per 512x32 B slab, issued by wave 0 only
  // (EXEC is ignored by tensor ops; one instruction per wave -> one DMA).
  auto stage_b = [&](int k1, int buf) {
    if (wv == 0)
      tdm_stage_b(lds_off(&sb[buf][0][0]),
                  (unsigned long long)(uintptr_t)w2b + (unsigned long long)(2 * k1));
  };

  // ---- pipeline prologue: fill buffer 0
  stage_b(0, 0);
  compute_h(0, 0);
  if (wv == 0) __builtin_amdgcn_s_wait_tensorcnt(0);
  __syncthreads();

  // ---- consumer wave mapping: 16 waves = 8 m-subtiles x 2 n-halves
  const int lane  = tid & 31;
  const int msub  = wv & 7;        // 16-row subtile
  const int nhalf = wv >> 3;       // 16 n-tiles each
  const int lrow  = lane & 15;
  const int hi    = lane >> 4;
  const int koffA = hi ? 8 : 0;    // A lanes 0-15: K{0..7,16..23}; 16-31: K{8..15,24..31}
  const int koffB = hi ? 8 : 0;    // B lanes 0-15: K{0..7,16..23}; 16-31: K{8..15,24..31}

  v8f acc[16] = {};

  for (int s = 0; s < NSTEPS; ++s) {
    const int buf = s & 1;
    if (s + 1 < NSTEPS) stage_b(K_STEP * (s + 1), buf ^ 1);   // TDM prefetch next B slab

    const unsigned short* ar = &sh_h[buf][msub * 16 + lrow][0];
    v8bf alo = *reinterpret_cast<const v8bf*>(ar + koffA);
    v8bf ahi = *reinterpret_cast<const v8bf*>(ar + 16 + koffA);
    v16bf a  = __builtin_shufflevector(alo, ahi,
                 0,1,2,3,4,5,6,7,8,9,10,11,12,13,14,15);

    // B fetch pipelined exactly one tile ahead; sched_barrier(0) stops the
    // scheduler from hoisting all 16 tiles' loads (which caused spills).
    const unsigned short* br0 = &sb[buf][(nhalf * 16) * 16 + lrow][koffB];
    v8bf blo = *reinterpret_cast<const v8bf*>(br0);
    v8bf bhi = *reinterpret_cast<const v8bf*>(br0 + 16);
    #pragma unroll
    for (int t = 0; t < 16; ++t) {
      v16bf b = __builtin_shufflevector(blo, bhi,
                  0,1,2,3,4,5,6,7,8,9,10,11,12,13,14,15);
      if (t + 1 < 16) {   // issue next tile's loads before this tile's WMMA
        const unsigned short* br = br0 + (size_t)(t + 1) * 16 * HPAD;
        blo = *reinterpret_cast<const v8bf*>(br);
        bhi = *reinterpret_cast<const v8bf*>(br + 16);
      }
      acc[t] = __builtin_amdgcn_wmma_f32_16x16x32_bf16(
                   false, a, false, b, (short)0, acc[t], false, false);
      __builtin_amdgcn_sched_barrier(0);
    }

    if (s + 1 < NSTEPS) {
      if (s + 2 < NSTEPS)
        __builtin_prefetch(w1 + (size_t)(K_STEP * (s + 2)) * NQ_DIM, 0, 2);
      compute_h(K_STEP * (s + 1), buf ^ 1);   // VALU co-executes with WMMA stream
    }

    if (wv == 0) __builtin_amdgcn_s_wait_tensorcnt(0);  // next B slab landed
    __syncthreads();                                    // publish to all waves
  }

  // ---- store: C layout -> VGPR j: M = j + 8*hi, N = lane%16 (64B coalesced runs)
  const int mbase = row0 + msub * 16 + hi * 8;
  #pragma unroll
  for (int t = 0; t < 16; ++t) {
    const int ncol = (nhalf * 16 + t) * 16 + lrow;
    #pragma unroll
    for (int j = 0; j < 8; ++j)
      out[(size_t)(mbase + j) * E_DIM + ncol] = acc[t][j];
  }
}

extern "C" void kernel_launch(void* const* d_in, const int* in_sizes, int n_in,
                              void* d_out, int out_size, void* d_ws, size_t ws_size,
                              hipStream_t stream) {
  const float* x   = (const float*)d_in[0];
  const float* rxs = (const float*)d_in[1];
  const float* w1  = (const float*)d_in[2];
  const float* w2  = (const float*)d_in[3];
  float* out = (float*)d_out;
  unsigned short* w2b = (unsigned short*)d_ws;   // 512*2048*2 = 2 MB

  const int n4 = (E_DIM * FFN_DIM) / 4;
  hipLaunchKernelGGL(cvt_w2_bf16, dim3((n4 + 255) / 256), dim3(256), 0, stream,
                     w2, w2b, n4);

  const int rows = 8 * 4096;                     // B*S = 32768
  hipLaunchKernelGGL(ffq_fused, dim3(rows / M_ROWS), dim3(THREADS), 0, stream,
                     x, rxs, w1, w2b, out);
}
